// S4DecisionTransformer_54958401519823
// MI455X (gfx1250) — matrix operand order repaired and checked by reading, not compiled, not run on hardware
//
#include <hip/hip_runtime.h>
#include <hip/hip_bf16.h>
#include <math.h>

typedef __attribute__((ext_vector_type(16))) _Float16 v16h;
typedef __attribute__((ext_vector_type(8)))  _Float16 v8h;
typedef __attribute__((ext_vector_type(8)))  float    v8f;

#define BB  16
#define TT  1024
#define HH  512
#define NN  32
#define SDD 128
#define ADD 32
#define LL  (3 * TT)          // 3072
#define ROWS (BB * LL)        // 49152
#define H2  (2 * HH)          // 1024

// ---------------------------------------------------------------------------
// 1) S4 discretization constants:  w = exp(dt*A),  C' = C*(w-1)/A
// ---------------------------------------------------------------------------
__global__ void s4_const_kernel(const float* __restrict__ log_dt,
                                const float* __restrict__ A_re,
                                const float* __restrict__ A_im,
                                const float* __restrict__ C_re,
                                const float* __restrict__ C_im,
                                float* __restrict__ w_re, float* __restrict__ w_im,
                                float* __restrict__ cp_re, float* __restrict__ cp_im) {
    int idx = blockIdx.x * blockDim.x + threadIdx.x;
    if (idx >= HH * NN) return;
    int h = idx / NN;
    float dt = __expf(log_dt[h]);
    float are = -__expf(A_re[idx]);
    float aim = A_im[idx];
    float dre = are * dt, dim = aim * dt;
    float ew  = __expf(dre);
    float wre = ew * __cosf(dim);
    float wim = ew * __sinf(dim);
    // (w-1)/A  via  (w-1)*conj(A)/|A|^2
    float m_re = wre - 1.0f, m_im = wim;
    float inv  = 1.0f / (are * are + aim * aim);
    float q_re = (m_re * are + m_im * aim) * inv;
    float q_im = (m_im * are - m_re * aim) * inv;
    float cr = C_re[idx], ci = C_im[idx];
    w_re[idx]  = wre;
    w_im[idx]  = wim;
    cp_re[idx] = cr * q_re - ci * q_im;
    cp_im[idx] = cr * q_im + ci * q_re;
}

// ---------------------------------------------------------------------------
// 2) W_out (2H,H) fp32 -> f16, transposed to K-major Bt[k][j] = W_out[j][k]
//    (1 MB, stays resident in L2/WGP$ for the WMMA GEMM)
// ---------------------------------------------------------------------------
__global__ void wout_prep_kernel(const float* __restrict__ W_out,
                                 _Float16* __restrict__ Bt) {
    int idx = blockIdx.x * blockDim.x + threadIdx.x;
    if (idx >= HH * H2) return;
    int k = idx / H2;          // input-dim (K)
    int j = idx % H2;          // output-dim (N)
    Bt[idx] = (_Float16)W_out[j * HH + k];
}

// ---------------------------------------------------------------------------
// 3) Embeddings + interleave (re,se,ae) + layernorm -> h16 (B*3T, H) f16
//    one workgroup per (b,t); 256 threads, 2 channels each
// ---------------------------------------------------------------------------
__device__ __forceinline__ float block_sum(float v, float* red, int tid) {
    red[tid] = v; __syncthreads();
    #pragma unroll
    for (int s = 128; s > 0; s >>= 1) {
        if (tid < s) red[tid] += red[tid + s];
        __syncthreads();
    }
    float r = red[0]; __syncthreads();
    return r;
}

__global__ void embed_ln_kernel(const int*   __restrict__ timesteps,
                                const float* __restrict__ states,
                                const float* __restrict__ actions,
                                const float* __restrict__ rtg,
                                const float* __restrict__ W_t,
                                const float* __restrict__ W_rtg,
                                const float* __restrict__ b_rtg,
                                const float* __restrict__ W_s,
                                const float* __restrict__ b_s,
                                const float* __restrict__ W_a,
                                const float* __restrict__ b_a,
                                const float* __restrict__ ln_g,
                                const float* __restrict__ ln_b,
                                _Float16* __restrict__ h16) {
    __shared__ float s_state[SDD];
    __shared__ float s_act[ADD];
    __shared__ float red[256];
    __shared__ float s_rtg;

    int bt  = blockIdx.x;                 // b*T + t
    int b   = bt / TT, t = bt % TT;
    int tid = threadIdx.x;

    if (tid < SDD) s_state[tid] = states[(size_t)bt * SDD + tid];
    if (tid < ADD) s_act[tid]   = actions[(size_t)bt * ADD + tid];
    if (tid == 0)  s_rtg        = rtg[bt];
    __syncthreads();

    int ts = timesteps[bt];
    float re_v[2], se_v[2], ae_v[2];
    #pragma unroll
    for (int j = 0; j < 2; ++j) {
        int c = tid + j * 256;
        float te = W_t[(size_t)ts * HH + c];
        float sa = b_s[c];
        #pragma unroll 8
        for (int k = 0; k < SDD; ++k) sa = fmaf(s_state[k], W_s[(size_t)c * SDD + k], sa);
        float aa = b_a[c];
        #pragma unroll 8
        for (int k = 0; k < ADD; ++k) aa = fmaf(s_act[k], W_a[(size_t)c * ADD + k], aa);
        se_v[j] = sa + te;
        ae_v[j] = aa + te;
        re_v[j] = b_rtg[c] + s_rtg * W_rtg[c] + te;
    }

    // layernorm each of the 3 H-vectors, write interleaved rows 3t+{0,1,2}
    size_t base_row = (size_t)b * LL + 3 * (size_t)t;
    float* vecs[3][2] = {{&re_v[0], &re_v[1]}, {&se_v[0], &se_v[1]}, {&ae_v[0], &ae_v[1]}};
    #pragma unroll
    for (int s = 0; s < 3; ++s) {
        float v0 = *vecs[s][0], v1 = *vecs[s][1];
        float sum = block_sum(v0 + v1, red, tid);
        float ssq = block_sum(v0 * v0 + v1 * v1, red, tid);
        float mean = sum * (1.0f / HH);
        float var  = ssq * (1.0f / HH) - mean * mean;
        float rstd = rsqrtf(var + 1e-5f);
        size_t row = base_row + s;
        #pragma unroll
        for (int j = 0; j < 2; ++j) {
            int c = tid + j * 256;
            float v = (j == 0) ? v0 : v1;
            h16[row * HH + c] = (_Float16)((v - mean) * rstd * ln_g[c] + ln_b[c]);
        }
    }
}

// ---------------------------------------------------------------------------
// 4) S4 diagonal scan: one wave per (b,h), one lane per complex state n.
//    x = w*x + u(t);  y(t) = 2*Re(sum_n C'_n x_n) + D_h*u(t);  gelu -> y16
//    32-step chunks with LDS 32x32 transpose for the cross-lane reduction.
// ---------------------------------------------------------------------------
__global__ void s4_scan_kernel(const _Float16* __restrict__ h16,
                               const float* __restrict__ w_re,
                               const float* __restrict__ w_im,
                               const float* __restrict__ cp_re,
                               const float* __restrict__ cp_im,
                               const float* __restrict__ Dvec,
                               _Float16* __restrict__ y16) {
    __shared__ float tile[8][32 * 33];      // per-wave 32x32 (+pad)

    int wave = threadIdx.x >> 5;
    int lane = threadIdx.x & 31;
    int chan = blockIdx.x * 8 + wave;       // b*H + h
    int b = chan / HH, h = chan % HH;

    int ci = h * NN + lane;
    float wre = w_re[ci], wim = w_im[ci];
    float cre = cp_re[ci], cim = cp_im[ci];
    float Dh = Dvec[h];
    float* tl = tile[wave];

    const _Float16* uptr = h16 + (size_t)b * LL * HH + h;
    _Float16*       yptr = y16 + (size_t)b * LL * HH + h;

    float xr = 0.0f, xi = 0.0f;
    for (int c = 0; c < LL / 32; ++c) {
        int t0 = c * 32;
        float u_val = (float)uptr[(size_t)(t0 + lane) * HH];
        // prefetch next chunk's strided u gather behind the recurrence
        if (c + 1 < LL / 32)
            __builtin_prefetch(uptr + (size_t)(t0 + 32 + lane) * HH, 0, 1);

        #pragma unroll 8
        for (int s = 0; s < 32; ++s) {
            float us = __shfl(u_val, s, 32);
            float nxr = fmaf(wre, xr, fmaf(-wim, xi, us));
            float nxi = fmaf(wre, xi, wim * xr);
            xr = nxr; xi = nxi;
            tl[s * 33 + lane] = cre * xr - cim * xi;
        }
        __syncthreads();

        // lane s owns timestep t0+s: reduce its row
        float acc = 0.0f;
        #pragma unroll 8
        for (int n = 0; n < 32; ++n) acc += tl[lane * 33 + n];
        float y = 2.0f * acc + Dh * u_val;
        // tanh-approx gelu (jax.nn.gelu default)
        float y3 = y * y * y;
        float g = 0.5f * y * (1.0f + tanhf(0.7978845608028654f * fmaf(0.044715f, y3, y)));
        yptr[(size_t)(t0 + lane) * HH] = (_Float16)g;
        __syncthreads();
    }
}

// ---------------------------------------------------------------------------
// 5) GLU output linear with WMMA f16 -> f32, GLU fused in epilogue.
//    z = gelu_y @ W_out.T + b_out ; out = z[:, :H] * sigmoid(z[:, H:])
//    Block = 8 waves over one 32-row stripe; wave w covers out cols [64w,64w+64)
//    and accumulates 'a' (col j) and 'b' (col j+512) tiles for TWO 16-row
//    tiles, reusing each B fragment across both -> 16 WMMA per K-step.
//    launch_bounds(.,1) + unroll(1) keep the 128-VGPR accumulator set
//    register-resident (round 2 spilled with full K-unroll).
// ---------------------------------------------------------------------------
__global__ void __launch_bounds__(256, 1)
glu_gemm_kernel(const _Float16* __restrict__ A,   // (ROWS, H) f16
                const _Float16* __restrict__ Bt,  // (H, 2H) f16, K-major
                const float* __restrict__ b_out,  // (2H)
                float* __restrict__ Out) {        // (ROWS, H) f32
    int wave = threadIdx.x >> 5;
    int lane = threadIdx.x & 31;
    int i0 = blockIdx.x * 32;
    int jbase = wave * 64;

    v8f acc_a[2][4], acc_b[2][4];
    #pragma unroll
    for (int rt = 0; rt < 2; ++rt)
        #pragma unroll
        for (int nt = 0; nt < 4; ++nt) { acc_a[rt][nt] = (v8f)0.0f; acc_b[rt][nt] = (v8f)0.0f; }

    const int m = lane & 15;
    const int abase = (lane >> 4) * 8;   // A-frag K interleave per ISA 7.12.2
    const _Float16* Arow0 = A + (size_t)(i0 + m) * HH;
    const _Float16* Arow1 = A + (size_t)(i0 + 16 + m) * HH;

    #pragma unroll 1
    for (int k0 = 0; k0 < HH; k0 += 32) {
        if (k0 + 32 < HH) {
            __builtin_prefetch(Arow0 + k0 + 32 + abase, 0, 1);
            __builtin_prefetch(Arow1 + k0 + 32 + abase, 0, 1);
        }
        v8h alo0 = *(const v8h*)(Arow0 + k0 + abase);        // K = abase..abase+7
        v8h ahi0 = *(const v8h*)(Arow0 + k0 + abase + 16);   // K = abase+16..+23
        v8h alo1 = *(const v8h*)(Arow1 + k0 + abase);
        v8h ahi1 = *(const v8h*)(Arow1 + k0 + abase + 16);
        v16h af0, af1;
        #pragma unroll
        for (int i = 0; i < 8; ++i) {
            af0[i] = alo0[i]; af0[8 + i] = ahi0[i];
            af1[i] = alo1[i]; af1[8 + i] = ahi1[i];
        }

        const _Float16* Bcol = Bt + (size_t)(k0 + abase) * H2 + jbase + m;  // N = lane&15, K = abase+s / abase+16+(s-8)
        #pragma unroll
        for (int nt = 0; nt < 4; ++nt) {
            v16h bfa, bfb;
            #pragma unroll
            for (int i = 0; i < 8; ++i) {
                bfa[i]     = Bcol[(size_t)i * H2 + nt * 16];
                bfa[8 + i] = Bcol[(size_t)(16 + i) * H2 + nt * 16];
                bfb[i]     = Bcol[(size_t)i * H2 + nt * 16 + HH];
                bfb[8 + i] = Bcol[(size_t)(16 + i) * H2 + nt * 16 + HH];
            }
            acc_a[0][nt] = __builtin_amdgcn_wmma_f32_16x16x32_f16(
                false, af0, false, bfa, (short)0, acc_a[0][nt], false, false);
            acc_a[1][nt] = __builtin_amdgcn_wmma_f32_16x16x32_f16(
                false, af1, false, bfa, (short)0, acc_a[1][nt], false, false);
            acc_b[0][nt] = __builtin_amdgcn_wmma_f32_16x16x32_f16(
                false, af0, false, bfb, (short)0, acc_b[0][nt], false, false);
            acc_b[1][nt] = __builtin_amdgcn_wmma_f32_16x16x32_f16(
                false, af1, false, bfb, (short)0, acc_b[1][nt], false, false);
        }
    }

    // epilogue: bias + GLU, D layout = lane(0-15): M=r, lane(16-31): M=r+8
    int rowoff = (lane >> 4) * 8;
    int ncol = lane & 15;
    #pragma unroll
    for (int nt = 0; nt < 4; ++nt) {
        int col = jbase + nt * 16 + ncol;
        float ba = b_out[col], bb = b_out[col + HH];
        #pragma unroll
        for (int rt = 0; rt < 2; ++rt) {
            #pragma unroll
            for (int r = 0; r < 8; ++r) {
                int row = i0 + rt * 16 + rowoff + r;
                float va = acc_a[rt][nt][r] + ba;
                float vb = acc_b[rt][nt][r] + bb;
                Out[(size_t)row * HH + col] = va / (1.0f + __expf(-vb));
            }
        }
    }
}

// ---------------------------------------------------------------------------
// 6) Prediction heads. One workgroup per (b,t): loads rows 3t+1 and 3t+2.
//    out layout: [state (B,T,128) | action (B,T,32) | rtg (B,T,1)]
// ---------------------------------------------------------------------------
__global__ void heads_kernel(const float* __restrict__ Hf,      // (B*3T, H)
                             const float* __restrict__ Wp_rtg, const float* __restrict__ bp_rtg,
                             const float* __restrict__ Wp_s,   const float* __restrict__ bp_s,
                             const float* __restrict__ Wp_a,   const float* __restrict__ bp_a,
                             float* __restrict__ out) {
    __shared__ float r2[HH];   // stream 2 (action positions)
    __shared__ float r1[HH];   // stream 1 (state positions)

    int bt = blockIdx.x;
    int b = bt / TT, t = bt % TT;
    int tid = threadIdx.x;

    size_t base = ((size_t)b * LL + 3 * (size_t)t) * HH;
    #pragma unroll
    for (int j = 0; j < 2; ++j) {
        int c = tid + j * 256;
        r1[c] = Hf[base + 1 * HH + c];
        r2[c] = Hf[base + 2 * HH + c];
    }
    __syncthreads();

    const size_t off_a = (size_t)BB * TT * SDD;             // 2097152
    const size_t off_r = off_a + (size_t)BB * TT * ADD;     // 2621440

    if (tid < SDD) {
        float acc = bp_s[tid];
        const float* w = Wp_s + (size_t)tid * HH;
        #pragma unroll 8
        for (int k = 0; k < HH; ++k) acc = fmaf(r2[k], w[k], acc);
        out[(size_t)bt * SDD + tid] = acc;
    } else if (tid < SDD + ADD) {
        int c = tid - SDD;
        float acc = bp_a[c];
        const float* w = Wp_a + (size_t)c * HH;
        #pragma unroll 8
        for (int k = 0; k < HH; ++k) acc = fmaf(r1[k], w[k], acc);
        out[off_a + (size_t)bt * ADD + c] = tanhf(acc);
    } else if (tid == SDD + ADD) {
        float acc = bp_rtg[0];
        #pragma unroll 8
        for (int k = 0; k < HH; ++k) acc = fmaf(r2[k], Wp_rtg[k], acc);
        out[off_r + (size_t)bt] = acc;
    }
}

// ---------------------------------------------------------------------------
extern "C" void kernel_launch(void* const* d_in, const int* in_sizes, int n_in,
                              void* d_out, int out_size, void* d_ws, size_t ws_size,
                              hipStream_t stream) {
    const int*   timesteps = (const int*)d_in[0];
    const float* states    = (const float*)d_in[1];
    const float* actions   = (const float*)d_in[2];
    const float* rtg       = (const float*)d_in[3];
    const float* W_t       = (const float*)d_in[4];
    const float* W_rtg     = (const float*)d_in[5];
    const float* b_rtg     = (const float*)d_in[6];
    const float* W_s       = (const float*)d_in[7];
    const float* b_s       = (const float*)d_in[8];
    const float* W_a       = (const float*)d_in[9];
    const float* b_a       = (const float*)d_in[10];
    const float* ln_g      = (const float*)d_in[11];
    const float* ln_b      = (const float*)d_in[12];
    const float* log_dt    = (const float*)d_in[13];
    const float* A_re      = (const float*)d_in[14];
    const float* A_im      = (const float*)d_in[15];
    const float* C_re      = (const float*)d_in[16];
    const float* C_im      = (const float*)d_in[17];
    const float* Dvec      = (const float*)d_in[18];
    const float* W_out     = (const float*)d_in[19];
    const float* b_out     = (const float*)d_in[20];
    const float* Wp_rtg    = (const float*)d_in[21];
    const float* bp_rtg    = (const float*)d_in[22];
    const float* Wp_s      = (const float*)d_in[23];
    const float* bp_s      = (const float*)d_in[24];
    const float* Wp_a      = (const float*)d_in[25];
    const float* bp_a      = (const float*)d_in[26];
    float* out = (float*)d_out;

    // workspace carve-out
    size_t off = 0;
    auto carve = [&](size_t bytes) -> void* {
        void* p = (char*)d_ws + off;
        off += (bytes + 255) & ~(size_t)255;
        return p;
    };
    _Float16* h16  = (_Float16*)carve((size_t)ROWS * HH * 2);   // 48 MB
    _Float16* y16  = (_Float16*)carve((size_t)ROWS * HH * 2);   // 48 MB
    float*    h2   = (float*)   carve((size_t)ROWS * HH * 4);   // 96 MB
    _Float16* wt16 = (_Float16*)carve((size_t)HH * H2 * 2);     // 1 MB
    float*    w_re = (float*)   carve((size_t)HH * NN * 4);
    float*    w_im = (float*)   carve((size_t)HH * NN * 4);
    float*    cpre = (float*)   carve((size_t)HH * NN * 4);
    float*    cpim = (float*)   carve((size_t)HH * NN * 4);

    s4_const_kernel<<<(HH * NN + 255) / 256, 256, 0, stream>>>(
        log_dt, A_re, A_im, C_re, C_im, w_re, w_im, cpre, cpim);

    wout_prep_kernel<<<(HH * H2 + 255) / 256, 256, 0, stream>>>(W_out, wt16);

    embed_ln_kernel<<<BB * TT, 256, 0, stream>>>(
        timesteps, states, actions, rtg, W_t, W_rtg, b_rtg,
        W_s, b_s, W_a, b_a, ln_g, ln_b, h16);

    s4_scan_kernel<<<(BB * HH) / 8, 256, 0, stream>>>(
        h16, w_re, w_im, cpre, cpim, Dvec, y16);

    glu_gemm_kernel<<<ROWS / 32, 256, 0, stream>>>(y16, wt16, b_out, h2);

    heads_kernel<<<BB * TT, 256, 0, stream>>>(
        h2, Wp_rtg, bp_rtg, Wp_s, bp_s, Wp_a, bp_a, out);
}
